// deform_up_17686675325482
// MI455X (gfx1250) — matrix operand order repaired and checked by reading, not compiled, not run on hardware
//
#include <hip/hip_runtime.h>
#include <math.h>

// ---------------------------------------------------------------------------
// deform-up block for MI455X (gfx1250, wave32, WMMA)
// convs = implicit GEMM on v_wmma_f32_16x16x32_f16 (f16 in, f32 accum)
//   A: weights pre-packed to WMMA fragment layout (f16)  -> 2x global b128/tap
//   B: LDS slab, channel-innermost (pitch 40 f16)        -> 2x ds b128/tap
// ---------------------------------------------------------------------------

typedef __attribute__((ext_vector_type(16))) _Float16 v16h;
typedef __attribute__((ext_vector_type(8)))  float    v8f;

#define NB     2
#define C1C    64
#define INCH   128
#define OUTCH  64
#define HH     256          // 2*H1
#define WW     256          // 2*W1
#define HW     (HH * WW)
#define BN_EPS 1e-5f

#define SLAB_PITCH 40                      // f16 per (row,col) cell (32 + pad)
#define SLAB_F16   (3 * 18 * SLAB_PITCH)   // 2160 f16 = 4320 B

// ---------------------------------------------------------------------------
// Kernel 0: pack conv weights (Cout,Cin,3,3) f32 -> WMMA A-fragment f16 layout
//   wp[(((ocTile*(Cin/32)+chunk)*9 + tap)*32 + lane)*16 + e]
//   lane: m = lane&15 (out-ch), h = lane>>4
//   e -> K = h*16 + e;  ic = chunk*32 + K
// ---------------------------------------------------------------------------
__global__ __launch_bounds__(256) void pack_weights_kernel(
    const float* __restrict__ w, _Float16* __restrict__ wp, int Cin, int Cout)
{
  size_t idx = (size_t)blockIdx.x * blockDim.x + threadIdx.x;  // < Cout*Cin*9
  int e    = (int)(idx & 15);
  int lane = (int)((idx >> 4) & 31);
  int rest = (int)(idx >> 9);            // (ocTile*(Cin/32)+chunk)*9 + tap
  int tap  = rest % 9;
  int tmp  = rest / 9;
  int nChunks = Cin / 32;
  int chunk  = tmp % nChunks;
  int ocTile = tmp / nChunks;
  int h  = lane >> 4;
  int oc = ocTile * 16 + (lane & 15);
  int K  = h * 16 + e;
  int ic = chunk * 32 + K;
  wp[idx] = (_Float16)w[((size_t)oc * Cin + ic) * 9 + tap];
}

// ---------------------------------------------------------------------------
// Kernel 1: ConvTranspose2d(k=2,s=2) + concat([x2, up]) -> xcat (NB,128,HH,WW)
// ---------------------------------------------------------------------------
__global__ __launch_bounds__(256) void upsample_concat_kernel(
    const float* __restrict__ x1,     // (NB,64,128,128)
    const float* __restrict__ x2,     // (NB,64,256,256)
    const float* __restrict__ up_w,   // (64,64,2,2)  torch IOHW
    const float* __restrict__ up_b,   // (64)
    float* __restrict__ xcat)         // (NB,128,256,256)
{
  size_t idx = (size_t)blockIdx.x * blockDim.x + threadIdx.x;   // < NB*128*HW
  int x = (int)(idx % WW);
  int y = (int)((idx / WW) % HH);
  int c = (int)((idx / HW) % INCH);
  int b = (int)(idx / ((size_t)HW * INCH));

  float val;
  if (c < C1C) {
    val = x2[((size_t)(b * C1C + c) * HH + y) * WW + x];
  } else {
    int o  = c - C1C;
    int hy = y >> 1, wx = x >> 1;
    int ky = y & 1,  kx = x & 1;
    float acc = up_b[o];
    const float* xp = x1 + ((size_t)b * C1C * 128 + hy) * 128 + wx;
    const float* wp = up_w + ((size_t)o * 2 + ky) * 2 + kx;
#pragma unroll 8
    for (int i = 0; i < C1C; ++i)
      acc += xp[(size_t)i * 128 * 128] * wp[(size_t)i * C1C * 4];
    val = acc;
  }
  xcat[idx] = val;
}

// ---------------------------------------------------------------------------
// Kernel 2: conv3x3 (pad=1) as implicit GEMM on v_wmma_f32_16x16x32_f16.
//  block = 128 threads = 4 waves; block covers 64 out-ch x 16 pixels.
//  Slab: [row][col][ic] channel-innermost, pitch 40 f16  -> B = 32B LDS read.
//  Weights: pre-packed fragments                         -> A = 32B global read.
// ---------------------------------------------------------------------------
__global__ __launch_bounds__(128) void conv3x3_wmma_kernel(
    const float* __restrict__ in,          // (NB,Cin,HH,WW) f32
    const _Float16* __restrict__ wpack,    // packed A fragments
    const float* __restrict__ bias,        // (Cout) or nullptr
    float* __restrict__ out,               // (NB,Cout,HH,WW)
    int Cin, int Cout)
{
  __shared__ __attribute__((aligned(32))) _Float16 slab[SLAB_F16];

  const int tile   = blockIdx.x;                 // HH * (WW/16) tiles
  const int y      = tile / (WW / 16);
  const int xBase  = (tile % (WW / 16)) * 16;
  const int b      = blockIdx.z;
  const int tid    = threadIdx.x;
  const int wave   = tid >> 5;
  const int lane   = tid & 31;
  const int n      = lane & 15;                  // column / pixel within tile
  const int h      = lane >> 4;                  // lane half
  const int ocTile = blockIdx.y * 4 + wave;      // 16-out-ch tile index
  const int nChunks = Cin / 32;

  v8f acc = {0.f, 0.f, 0.f, 0.f, 0.f, 0.f, 0.f, 0.f};

  for (int chunk = 0; chunk < nChunks; ++chunk) {
    const int icBase = chunk * 32;
    __syncthreads();
    // cooperative slab fill: 32ch x 3rows x 18cols, col-innermost decode so
    // global reads coalesce; f32 -> f16; zero pad outside the image.
    for (int t = tid; t < 32 * 3 * 18; t += 128) {
      int col = t % 18;
      int ic  = (t / 18) % 32;
      int row = t / (18 * 32);
      int gy = y + row - 1;
      int gx = xBase + col - 1;
      float v = 0.f;
      if (gy >= 0 && gy < HH && gx >= 0 && gx < WW)
        v = in[(((size_t)b * Cin + icBase + ic) * HH + gy) * WW + gx];
      slab[(row * 18 + col) * SLAB_PITCH + ic] = (_Float16)v;
    }
    // prefetch next chunk's center rows into cache (global_prefetch_b8)
    if (icBase + 32 < Cin) {
      const float* nxt =
          in + (((size_t)b * Cin + icBase + 32 + lane) * HH + y) * WW + xBase;
      __builtin_prefetch(nxt, 0, 1);
    }
    __syncthreads();

    const _Float16* apBase =
        wpack + ((((size_t)ocTile * nChunks + chunk) * 9) * 32 + lane) * 16;

#pragma unroll
    for (int tap = 0; tap < 9; ++tap) {
      const int dy = tap / 3, dx = tap % 3;
      // B operand (32x16 f16): lane half h holds K = h*16+e, col = n
      v16h Bv = *(const v16h*)&slab[((dy * 18) + n + dx) * SLAB_PITCH + h * 16];
      // A operand: packed fragment, contiguous 32B per lane
      v16h Av = *(const v16h*)(apBase + (size_t)tap * 32 * 16);

      acc = __builtin_amdgcn_wmma_f32_16x16x32_f16(
          /*neg_a=*/false, Av, /*neg_b=*/false, Bv,
          /*c_mod=*/(short)0, acc, /*reuse_a=*/false, /*reuse_b=*/false);
    }
  }

  // C/D layout: VGPR r, lane half h -> element (M = r + 8*h, N = n)
#pragma unroll
  for (int r = 0; r < 8; ++r) {
    int oc = ocTile * 16 + r + h * 8;
    float v = acc[r];
    if (bias) v += bias[oc];
    out[(((size_t)b * Cout + oc) * HH + y) * WW + xBase + n] = v;
  }
}

// ---------------------------------------------------------------------------
// Kernel 3: deformable bilinear gather (faithful flat-reinterpret of offsets)
// ---------------------------------------------------------------------------
__global__ __launch_bounds__(256) void deform_gather_kernel(
    const float* __restrict__ xcat,     // (NB*128, HH, WW) flat view
    const float* __restrict__ offsets,  // (NB,256,HH,WW) flat
    float* __restrict__ xd)             // (NB*128, HH, WW)
{
  size_t idx = (size_t)blockIdx.x * blockDim.x + threadIdx.x;   // < NB*128*HW
  size_t bc  = idx / HW;
  int pix = (int)(idx % HW);
  int y = pix / WW;
  int x = pix % WW;

  float offy = offsets[idx * 2 + 0];
  float offx = offsets[idx * 2 + 1];

  float cy = fminf(fmaxf(offy + (float)y, 0.f), (float)(HH - 1));
  float cx = fminf(fmaxf(offx + (float)x, 0.f), (float)(WW - 1));
  float y0f = floorf(cy), x0f = floorf(cx);
  int y0 = (int)y0f,        x0 = (int)x0f;
  int y1 = (int)ceilf(cy),  x1 = (int)ceilf(cx);

  const float* plane = xcat + bc * HW;
  float v00 = plane[y0 * WW + x0];
  float v10 = plane[y1 * WW + x0];
  float v01 = plane[y0 * WW + x1];
  float v11 = plane[y1 * WW + x1];
  float wy = cy - y0f, wx = cx - x0f;
  float vt = v00 + (v10 - v00) * wy;
  float vb = v01 + (v11 - v01) * wy;
  xd[idx] = vt + (vb - vt) * wx;
}

// ---------------------------------------------------------------------------
// Kernel 4: per-channel sum / sumsq over (N,H,W)  (one block per channel)
// ---------------------------------------------------------------------------
__global__ __launch_bounds__(256) void bn_stats_kernel(
    const float* __restrict__ in, int C,
    float* __restrict__ sum, float* __restrict__ sumsq)
{
  __shared__ float rs[256];
  __shared__ float rss[256];
  const int c = blockIdx.x;
  const int tid = threadIdx.x;
  const int NTOT = NB * HW;

  float s = 0.f, ss = 0.f;
  for (int i = tid; i < NTOT; i += 256) {
    int b = i / HW;
    int pix = i % HW;
    float v = in[((size_t)b * C + c) * HW + pix];
    s += v;
    ss += v * v;
  }
  rs[tid] = s; rss[tid] = ss;
  __syncthreads();
  for (int o = 128; o > 0; o >>= 1) {
    if (tid < o) { rs[tid] += rs[tid + o]; rss[tid] += rss[tid + o]; }
    __syncthreads();
  }
  if (tid == 0) { sum[c] = rs[0]; sumsq[c] = rss[0]; }
}

// ---------------------------------------------------------------------------
// Kernel 5: BatchNorm (batch stats, biased var) + ReLU, elementwise
// ---------------------------------------------------------------------------
__global__ __launch_bounds__(256) void bn_relu_kernel(
    const float* __restrict__ in, int C,
    const float* __restrict__ sum, const float* __restrict__ sumsq,
    const float* __restrict__ g, const float* __restrict__ bb,
    float* __restrict__ out)
{
  size_t idx = (size_t)blockIdx.x * blockDim.x + threadIdx.x;   // < NB*C*HW
  int c = (int)((idx / HW) % C);
  const float invN = 1.f / (float)(NB * HW);
  float m = sum[c] * invN;
  float var = sumsq[c] * invN - m * m;
  float v = (in[idx] - m) * rsqrtf(var + BN_EPS) * g[c] + bb[c];
  out[idx] = v > 0.f ? v : 0.f;
}

// ---------------------------------------------------------------------------
extern "C" void kernel_launch(void* const* d_in, const int* in_sizes, int n_in,
                              void* d_out, int out_size, void* d_ws, size_t ws_size,
                              hipStream_t stream)
{
  const float* x1   = (const float*)d_in[0];
  const float* x2   = (const float*)d_in[1];
  const float* up_w = (const float*)d_in[2];
  const float* up_b = (const float*)d_in[3];
  const float* off_w= (const float*)d_in[4];
  const float* c1_w = (const float*)d_in[5];
  const float* c1_b = (const float*)d_in[6];
  const float* g1   = (const float*)d_in[7];
  const float* b1   = (const float*)d_in[8];
  const float* c2_w = (const float*)d_in[9];
  const float* c2_b = (const float*)d_in[10];
  const float* g2   = (const float*)d_in[11];
  const float* b2   = (const float*)d_in[12];
  float* outp = (float*)d_out;

  // workspace layout
  const size_t xcat_elems = (size_t)NB * INCH * HW;      // 16.78M f32
  const size_t off_elems  = (size_t)NB * 2 * INCH * HW;  // 33.55M f32
  const size_t y_elems    = (size_t)NB * OUTCH * HW;     //  8.39M f32

  float* xcat    = (float*)d_ws;                         // [0, 64 MiB)
  float* offsets = xcat + xcat_elems;                    // [64, 192 MiB)
  float* xd      = offsets + off_elems;                  // [192, 256 MiB)
  float* y1      = offsets;                              // reuse after gather
  float* hbuf    = offsets + y_elems;
  float* y2      = offsets + 2 * y_elems;
  float* stats   = xd + xcat_elems;                      // 512 f32 (32B-aligned)
  float* sum_c   = stats;
  float* ssq_c   = stats + 256;

  // packed f16 weight fragments (32B-aligned: all region sizes are mult. of 32B)
  const size_t woff_elems = (size_t)(2 * INCH) * INCH * 9;   // 294912 f16
  const size_t wc1_elems  = (size_t)OUTCH * INCH * 9;        //  73728 f16
  const size_t wc2_elems  = (size_t)OUTCH * OUTCH * 9;       //  36864 f16
  _Float16* wpk_off = (_Float16*)(stats + 512);
  _Float16* wpk_c1  = wpk_off + woff_elems;
  _Float16* wpk_c2  = wpk_c1 + wc1_elems;

  // 0) pack weights into WMMA fragment layout (tiny, one pass each)
  pack_weights_kernel<<<(unsigned)(woff_elems / 256), 256, 0, stream>>>(
      off_w, wpk_off, INCH, 2 * INCH);
  pack_weights_kernel<<<(unsigned)(wc1_elems / 256), 256, 0, stream>>>(
      c1_w, wpk_c1, INCH, OUTCH);
  pack_weights_kernel<<<(unsigned)(wc2_elems / 256), 256, 0, stream>>>(
      c2_w, wpk_c2, OUTCH, OUTCH);

  // 1) upsample + concat
  upsample_concat_kernel<<<(unsigned)(xcat_elems / 256), 256, 0, stream>>>(
      x1, x2, up_w, up_b, xcat);

  // 2) offsets = conv3x3(xcat, off_w), 128 -> 256 ch, no bias
  conv3x3_wmma_kernel<<<dim3(HH * (WW / 16), (2 * INCH) / 64, NB), 128, 0, stream>>>(
      xcat, wpk_off, nullptr, offsets, INCH, 2 * INCH);

  // 3) deformable bilinear gather -> xd
  deform_gather_kernel<<<(unsigned)(xcat_elems / 256), 256, 0, stream>>>(
      xcat, offsets, xd);

  // 4) conv1 (128 -> 64) + BN + ReLU
  conv3x3_wmma_kernel<<<dim3(HH * (WW / 16), OUTCH / 64, NB), 128, 0, stream>>>(
      xd, wpk_c1, c1_b, y1, INCH, OUTCH);
  bn_stats_kernel<<<OUTCH, 256, 0, stream>>>(y1, OUTCH, sum_c, ssq_c);
  bn_relu_kernel<<<(unsigned)(y_elems / 256), 256, 0, stream>>>(
      y1, OUTCH, sum_c, ssq_c, g1, b1, hbuf);

  // 5) conv2 (64 -> 64) + BN + ReLU -> d_out
  conv3x3_wmma_kernel<<<dim3(HH * (WW / 16), OUTCH / 64, NB), 128, 0, stream>>>(
      hbuf, wpk_c2, c2_b, y2, OUTCH, OUTCH);
  bn_stats_kernel<<<OUTCH, 256, 0, stream>>>(y2, OUTCH, sum_c, ssq_c);
  bn_relu_kernel<<<(unsigned)(y_elems / 256), 256, 0, stream>>>(
      y2, OUTCH, sum_c, ssq_c, g2, b2, outp);

  (void)in_sizes; (void)n_in; (void)out_size; (void)ws_size;
}
